// MH_Lori_24163486007330
// MI455X (gfx1250) — matrix operand order repaired and checked by reading, not compiled, not run on hardware
//
#include <hip/hip_runtime.h>

// ---------------------------------------------------------------------------
// MI455X (gfx1250): bf16 WMMA everywhere, f32 accumulate.
//   1) convert x, mh_W, merge_W, first_linear, second_linear -> bf16 (ws)
//   2) GEMM1: x1 = x @ mh_W.T + mh_b   (async-LDS double-buffered WMMA GEMM)
//   3) router: segment means -> softmax weights
//   4) FFN: per (b,n,h) merge expert weights in LDS chunks, two WMMA GEMMs
//   5) GEMM4: out = res @ merge_W.T + merge_b  (same GEMM, f32 epilogue)
// ---------------------------------------------------------------------------

typedef __bf16 bf16_t;
typedef __attribute__((ext_vector_type(16))) __bf16        v16bf;
typedef __attribute__((ext_vector_type(8)))  float         v8f;
typedef __attribute__((ext_vector_type(4)))  unsigned int  v4u;

#define B_   4
#define S_   2048
#define HD_  1024
#define H_   8
#define D_   128
#define E_   8
#define I_   512
#define N_   16
#define L_   128
#define M_   (B_ * S_)          // 8192 rows

#define LDT  40                 // halves/row for 32-K LDS chunks (bank-safe, 16B mult)
#define LDH  136                // halves/row for 128-wide LDS tiles (bank-safe, 16B mult)
#define GBUF_BYTES (128 * LDT * 2)   // one staging buffer = 10240 B

union Frag16 { v16bf h; v4u q[2]; };
union H8     { v4u q; bf16_t h[8]; };

// A-matrix 16x32 bf16 fragment from row-major [16 x ldh] at `base`.
// Lanes 0-15 hold K=0..7,16..23 of row M=lane; lanes 16-31 hold K=8..15,24..31.
__device__ __forceinline__ v16bf ld_frag_a(const bf16_t* base, int ldh, int lane) {
  const int m = lane & 15, sel = lane >> 4;
  const bf16_t* r = base + m * ldh + sel * 8;
  Frag16 f;
  f.q[0] = *(const v4u*)(r);
  f.q[1] = *(const v4u*)(r + 16);
  return f.h;
}

// B-matrix 32x16 bf16 fragment; weight row-major [N x K] -> lane n holds
// W[n, sel*8 .. sel*8+7] and W[n, 16+sel*8 .. 16+sel*8+7] (same K order as A).
__device__ __forceinline__ v16bf ld_frag_b(const bf16_t* base, int ldh, int lane) {
  const int n = lane & 15, sel = lane >> 4;
  const bf16_t* r = base + n * ldh + sel * 8;
  Frag16 f;
  f.q[0] = *(const v4u*)(r);
  f.q[1] = *(const v4u*)(r + 16);
  return f.h;
}

__device__ __forceinline__ v8f wmma_bf16(v16bf a, v16bf b, v8f c) {
  return __builtin_amdgcn_wmma_f32_16x16x32_bf16(false, a, false, b, (short)0, c,
                                                 false, false);
}

__device__ __forceinline__ v8f v8f_zero() {
  v8f z = {0.f, 0.f, 0.f, 0.f, 0.f, 0.f, 0.f, 0.f};
  return z;
}

// CDNA5 async HBM->LDS copy (ASYNCcnt-tracked, no VGPR round trip).
// lds = wave-relative LDS byte address (generic ptr truncated to 32 bits).
__device__ __forceinline__ void async_copy_b128(unsigned lds, const bf16_t* gp) {
  asm volatile("global_load_async_to_lds_b128 %0, %1, off"
               :: "v"(lds), "v"(gp) : "memory");
}
__device__ __forceinline__ void wait_async0() {
  asm volatile("s_wait_asynccnt 0x0" ::: "memory");
}

// ---------------------------------------------------------------------------
__global__ void cvt_f32_bf16(const float* __restrict__ in, bf16_t* __restrict__ out,
                             int n) {
  int i = blockIdx.x * blockDim.x + threadIdx.x;
  if (i < n) out[i] = (bf16_t)in[i];
}

// ---------------------------------------------------------------------------
// C[m,n] = sum_k A[m,k] * Bw[n,k] + bias[n]    (torch Linear)
// A: [Mdim x Kdim] bf16, Bw: [Ndim x Kdim] bf16.
// Block = 256 thr (8 waves), macro-tile 128x128, wave tile 32x64, K step 32.
// Double-buffered LDS staged by global_load_async_to_lds_b128.
template <bool WRITE_F32>
__global__ __launch_bounds__(256)
void gemm_bf16(const bf16_t* __restrict__ A, const bf16_t* __restrict__ Bw,
               const float* __restrict__ bias, void* __restrict__ Cout,
               int Mdim, int Ndim, int Kdim) {
  __shared__ __align__(16) bf16_t sA[2][128 * LDT];
  __shared__ __align__(16) bf16_t sB[2][128 * LDT];

  const int tid  = threadIdx.x;
  const int lane = tid & 31;
  const int wave = tid >> 5;
  const int wr   = wave >> 1;          // 0..3 : row group of 32
  const int wc   = wave & 1;           // 0..1 : col group of 64
  const int bm   = blockIdx.y * 128;
  const int bn   = blockIdx.x * 128;

  // staging: 512 assignments (128 rows x 4 16B-chunks), 2 per thread, per array
  const int a0 = tid, a1 = tid + 256;
  const int r0 = a0 >> 2, c0 = a0 & 3;
  const int r1 = a1 >> 2, c1 = a1 & 3;
  const bf16_t* gA0 = &A [(size_t)(bm + r0) * Kdim + c0 * 8];
  const bf16_t* gA1 = &A [(size_t)(bm + r1) * Kdim + c1 * 8];
  const bf16_t* gB0 = &Bw[(size_t)(bn + r0) * Kdim + c0 * 8];
  const bf16_t* gB1 = &Bw[(size_t)(bn + r1) * Kdim + c1 * 8];
  const unsigned lA0 = (unsigned)(size_t)&sA[0][r0 * LDT + c0 * 8];
  const unsigned lA1 = (unsigned)(size_t)&sA[0][r1 * LDT + c1 * 8];
  const unsigned lB0 = (unsigned)(size_t)&sB[0][r0 * LDT + c0 * 8];
  const unsigned lB1 = (unsigned)(size_t)&sB[0][r1 * LDT + c1 * 8];

  auto issue = [&](int k, int p) {
    const size_t ko = (size_t)k * 32;
    const unsigned o = (unsigned)p * GBUF_BYTES;
    async_copy_b128(lA0 + o, gA0 + ko);
    async_copy_b128(lA1 + o, gA1 + ko);
    async_copy_b128(lB0 + o, gB0 + ko);
    async_copy_b128(lB1 + o, gB1 + ko);
  };

  v8f acc[2][4];
#pragma unroll
  for (int r = 0; r < 2; ++r)
#pragma unroll
    for (int c = 0; c < 4; ++c) acc[r][c] = v8f_zero();

  const int nk = Kdim / 32;
  issue(0, 0);
  for (int k = 0; k < nk; ++k) {
    const int p = k & 1;
    wait_async0();        // this wave's portion of chunk k landed in LDS
    __syncthreads();      // all waves' portions visible; prev reads of buf p^1 done
    if (k + 1 < nk) issue(k + 1, p ^ 1);

    const bf16_t* cA = sA[p];
    const bf16_t* cB = sB[p];
    v16bf af[2], bfr[4];
    af[0] = ld_frag_a(&cA[(wr * 32 + 0)  * LDT], LDT, lane);
    af[1] = ld_frag_a(&cA[(wr * 32 + 16) * LDT], LDT, lane);
#pragma unroll
    for (int c = 0; c < 4; ++c)
      bfr[c] = ld_frag_b(&cB[(wc * 64 + c * 16) * LDT], LDT, lane);
#pragma unroll
    for (int r = 0; r < 2; ++r)
#pragma unroll
      for (int c = 0; c < 4; ++c) acc[r][c] = wmma_bf16(af[r], bfr[c], acc[r][c]);
  }

  // epilogue: D layout -> lane holds n = lane&15, rows m = sel*8 + v
  const int nn = lane & 15, sel = lane >> 4;
#pragma unroll
  for (int r = 0; r < 2; ++r)
#pragma unroll
    for (int c = 0; c < 4; ++c) {
      int gm = bm + wr * 32 + r * 16 + sel * 8;
      int gn = bn + wc * 64 + c * 16 + nn;
      float bv = bias[gn];
#pragma unroll
      for (int v = 0; v < 8; ++v) {
        float val = acc[r][c][v] + bv;
        size_t idx = (size_t)(gm + v) * Ndim + gn;
        if (WRITE_F32) ((float*)Cout)[idx] = val;
        else           ((bf16_t*)Cout)[idx] = (bf16_t)val;
      }
    }
}

// ---------------------------------------------------------------------------
// One block per (b,n): mean over L of x1 segment, logits vs expert_emb,
// softmax over E. Output wrt[b,n,h,e].
__global__ __launch_bounds__(256)
void router_kernel(const bf16_t* __restrict__ x1, const float* __restrict__ emb,
                   float* __restrict__ wrt) {
  __shared__ float avg[HD_];
  __shared__ float logits[H_ * E_];
  const int bn = blockIdx.x;                       // b*N + n
  const size_t rowbase = (size_t)bn * L_ * HD_;

  for (int j = threadIdx.x; j < HD_; j += 256) {
    float s = 0.f;
    for (int l = 0; l < L_; ++l) s += (float)x1[rowbase + (size_t)l * HD_ + j];
    avg[j] = s * (1.0f / L_);
  }
  __syncthreads();

  if (threadIdx.x < H_ * E_) {
    int h = threadIdx.x >> 3, e = threadIdx.x & 7;
    float s = 0.f;
    for (int dd = 0; dd < D_; ++dd) s += avg[h * D_ + dd] * emb[dd * E_ + e];
    logits[threadIdx.x] = s;
  }
  __syncthreads();

  if (threadIdx.x < H_) {
    int h = threadIdx.x;
    float mx = -1e30f;
    for (int e = 0; e < E_; ++e) mx = fmaxf(mx, logits[h * E_ + e]);
    float ex[E_], sum = 0.f;
    for (int e = 0; e < E_; ++e) { ex[e] = __expf(logits[h * E_ + e] - mx); sum += ex[e]; }
    float inv = 1.f / sum;
    for (int e = 0; e < E_; ++e)
      wrt[((size_t)bn * H_ + h) * E_ + e] = ex[e] * inv;
  }
}

// ---------------------------------------------------------------------------
// One block per (b,n,h): merged-expert FFN over its 128x128 segment slice.
//   hc  = relu(xs @ m1.T)   xs:[L,d]  m1:[I,d] merged (chunked in LDS)
//   out = hc @ m2.T          m2:[d,I] merged (chunked in LDS)
// Causal: weights from segment (n==0 ? 0 : n-1).
__global__ __launch_bounds__(256)
void ffn_kernel(const bf16_t* __restrict__ x1, const bf16_t* __restrict__ fl,
                const bf16_t* __restrict__ sl, const float* __restrict__ wrt,
                bf16_t* __restrict__ res) {
  __shared__ __align__(16) bf16_t sW[128 * LDT];   // merged weight 32-K chunk
  __shared__ __align__(16) bf16_t sH[128 * LDH];   // hc = relu(first gemm), bf16
  __shared__ float sw[E_];

  const int blk = blockIdx.x;                      // (b*N + n)*H + h
  const int h   = blk % H_;
  const int n   = (blk / H_) % N_;
  const int b   = blk / (H_ * N_);
  const int nw  = (n == 0) ? 0 : n - 1;

  if (threadIdx.x < E_)
    sw[threadIdx.x] = wrt[(((size_t)(b * N_ + nw)) * H_ + h) * E_ + threadIdx.x];
  __syncthreads();

  float wreg[E_];
#pragma unroll
  for (int e = 0; e < E_; ++e) wreg[e] = sw[e];

  const int lane = threadIdx.x & 31;
  const int wave = threadIdx.x >> 5;
  const int wr   = wave >> 1, wc = wave & 1;
  const int nn   = lane & 15, sel = lane >> 4;

  const size_t xbase = ((size_t)(b * N_ + n) * L_) * HD_ + h * D_;

  v8f oacc[2][4];
#pragma unroll
  for (int r = 0; r < 2; ++r)
#pragma unroll
    for (int c = 0; c < 4; ++c) oacc[r][c] = v8f_zero();

  for (int ic = 0; ic < 4; ++ic) {                 // I chunks of 128
    // ---------- first GEMM: hacc[l, i_local] over k=0..127 ----------
    v8f hacc[2][4];
#pragma unroll
    for (int r = 0; r < 2; ++r)
#pragma unroll
      for (int c = 0; c < 4; ++c) hacc[r][c] = v8f_zero();

    for (int kc = 0; kc < 4; ++kc) {
      __syncthreads();                             // prior sW readers done
      // build m1 chunk: sW[i_local, kk] = sum_e w[e]*W1[e, ic*128+i_local, kc*32+kk]
#pragma unroll
      for (int it = 0; it < 2; ++it) {
        int a = threadIdx.x + it * 256;
        int row = a >> 2, cc = a & 3;
        float av[8];
#pragma unroll
        for (int j = 0; j < 8; ++j) av[j] = 0.f;
#pragma unroll
        for (int e = 0; e < E_; ++e) {
          H8 u;
          u.q = *(const v4u*)(&fl[((size_t)e * I_ + ic * 128 + row) * D_ +
                                  kc * 32 + cc * 8]);
#pragma unroll
          for (int j = 0; j < 8; ++j) av[j] += wreg[e] * (float)u.h[j];
        }
        H8 o;
#pragma unroll
        for (int j = 0; j < 8; ++j) o.h[j] = (bf16_t)av[j];
        *(v4u*)(&sW[row * LDT + cc * 8]) = o.q;
      }
      __syncthreads();

      v16bf af[2], bfr[4];
      af[0] = ld_frag_a(&x1[xbase + (size_t)(wr * 32 + 0)  * HD_ + kc * 32], HD_, lane);
      af[1] = ld_frag_a(&x1[xbase + (size_t)(wr * 32 + 16) * HD_ + kc * 32], HD_, lane);
#pragma unroll
      for (int c = 0; c < 4; ++c)
        bfr[c] = ld_frag_b(&sW[(wc * 64 + c * 16) * LDT], LDT, lane);
#pragma unroll
      for (int r = 0; r < 2; ++r)
#pragma unroll
        for (int c = 0; c < 4; ++c) hacc[r][c] = wmma_bf16(af[r], bfr[c], hacc[r][c]);
    }

    __syncthreads();                               // prior sH readers done
    // relu + pack hc into LDS as A-matrix for second gemm
#pragma unroll
    for (int r = 0; r < 2; ++r)
#pragma unroll
      for (int c = 0; c < 4; ++c)
#pragma unroll
        for (int v = 0; v < 8; ++v) {
          float x = hacc[r][c][v];
          x = x > 0.f ? x : 0.f;
          int l = wr * 32 + r * 16 + sel * 8 + v;
          int i = wc * 64 + c * 16 + nn;
          sH[l * LDH + i] = (bf16_t)x;
        }
    __syncthreads();

    // ---------- second GEMM: oacc[l, dd] += hc @ m2c.T over this I chunk ----------
    for (int kc = 0; kc < 4; ++kc) {
      __syncthreads();
      // build m2 chunk: sW[dd, ii] = sum_e w[e]*W2[e, dd, ic*128+kc*32+ii]
#pragma unroll
      for (int it = 0; it < 2; ++it) {
        int a = threadIdx.x + it * 256;
        int row = a >> 2, cc = a & 3;
        float av[8];
#pragma unroll
        for (int j = 0; j < 8; ++j) av[j] = 0.f;
#pragma unroll
        for (int e = 0; e < E_; ++e) {
          H8 u;
          u.q = *(const v4u*)(&sl[((size_t)e * D_ + row) * I_ + ic * 128 +
                                  kc * 32 + cc * 8]);
#pragma unroll
          for (int j = 0; j < 8; ++j) av[j] += wreg[e] * (float)u.h[j];
        }
        H8 o;
#pragma unroll
        for (int j = 0; j < 8; ++j) o.h[j] = (bf16_t)av[j];
        *(v4u*)(&sW[row * LDT + cc * 8]) = o.q;
      }
      __syncthreads();

      v16bf af[2], bfr[4];
      af[0] = ld_frag_a(&sH[(wr * 32 + 0)  * LDH + kc * 32], LDH, lane);
      af[1] = ld_frag_a(&sH[(wr * 32 + 16) * LDH + kc * 32], LDH, lane);
#pragma unroll
      for (int c = 0; c < 4; ++c)
        bfr[c] = ld_frag_b(&sW[(wc * 64 + c * 16) * LDT], LDT, lane);
#pragma unroll
      for (int r = 0; r < 2; ++r)
#pragma unroll
        for (int c = 0; c < 4; ++c) oacc[r][c] = wmma_bf16(af[r], bfr[c], oacc[r][c]);
    }
  }

  // write res (bf16) at [B,S,Hd] layout: row b*S + n*L + l, col h*128 + dd
  const size_t obase = ((size_t)(b * N_ + n) * L_) * HD_ + h * D_;
#pragma unroll
  for (int r = 0; r < 2; ++r)
#pragma unroll
    for (int c = 0; c < 4; ++c)
#pragma unroll
      for (int v = 0; v < 8; ++v) {
        int l  = wr * 32 + r * 16 + sel * 8 + v;
        int dd = wc * 64 + c * 16 + nn;
        res[obase + (size_t)l * HD_ + dd] = (bf16_t)oacc[r][c][v];
      }
}

// ---------------------------------------------------------------------------
extern "C" void kernel_launch(void* const* d_in, const int* in_sizes, int n_in,
                              void* d_out, int out_size, void* d_ws, size_t ws_size,
                              hipStream_t stream) {
  const float* x   = (const float*)d_in[0];
  const float* mhW = (const float*)d_in[1];
  const float* mhb = (const float*)d_in[2];
  const float* emb = (const float*)d_in[3];
  const float* fl  = (const float*)d_in[4];
  const float* sl  = (const float*)d_in[5];
  const float* mW  = (const float*)d_in[6];
  const float* mb  = (const float*)d_in[7];
  float* out = (float*)d_out;

  // workspace layout (bf16 halves); res aliases xb (xb only needed by GEMM1)
  bf16_t* xb   = (bf16_t*)d_ws;                        // M*HD
  bf16_t* resb = xb;                                   // alias (sequential use)
  bf16_t* x1b  = xb   + (size_t)M_ * HD_;              // M*HD
  bf16_t* mhWb = x1b  + (size_t)M_ * HD_;              // HD*HD
  bf16_t* mWb  = mhWb + (size_t)HD_ * HD_;             // HD*HD
  bf16_t* flb  = mWb  + (size_t)HD_ * HD_;             // E*I*D
  bf16_t* slb  = flb  + (size_t)E_ * I_ * D_;          // E*D*I
  float*  wrt  = (float*)(slb + (size_t)E_ * I_ * D_); // B*N*H*E

  auto cvt = [&](const float* src, bf16_t* dst, int n) {
    cvt_f32_bf16<<<(n + 255) / 256, 256, 0, stream>>>(src, dst, n);
  };
  cvt(x,   xb,   M_ * HD_);
  cvt(mhW, mhWb, HD_ * HD_);
  cvt(mW,  mWb,  HD_ * HD_);
  cvt(fl,  flb,  E_ * I_ * D_);
  cvt(sl,  slb,  E_ * D_ * I_);

  dim3 ggrid(HD_ / 128, M_ / 128);  // (8, 64)

  // x1 = x @ mh_W.T + mh_b  -> bf16
  gemm_bf16<false><<<ggrid, 256, 0, stream>>>(xb, mhWb, mhb, (void*)x1b,
                                              M_, HD_, HD_);

  // router softmax weights
  router_kernel<<<B_ * N_, 256, 0, stream>>>(x1b, emb, wrt);

  // merged-expert FFN -> res (bf16)
  ffn_kernel<<<B_ * N_ * H_, 256, 0, stream>>>(x1b, flb, slb, wrt, resb);

  // out = res @ merge_W.T + merge_b -> f32
  gemm_bf16<true><<<ggrid, 256, 0, stream>>>(resb, mWb, mb, (void*)out,
                                             M_, HD_, HD_);

  (void)in_sizes; (void)n_in; (void)out_size; (void)ws_size;
}
